// HGPN_30142080484001
// MI455X (gfx1250) — matrix operand order repaired and checked by reading, compiled, not run on hardware
//
#include <hip/hip_runtime.h>
#include <stdint.h>

// ---------------------------------------------------------------------------
// CDNA5 (gfx1250) implementation. wave32, WMMA bf16 (f32 accum).
// ---------------------------------------------------------------------------

#define DEV __device__ __forceinline__

typedef __attribute__((ext_vector_type(16))) __bf16 bf16x16;
typedef __attribute__((ext_vector_type(8)))  float  f32x8;

union FragU { uint32_t u[8]; bf16x16 v; };

DEV uint32_t f2bf_bits(float f) {
    union { float f; uint32_t u; } x; x.f = f;
    uint32_t u = x.u;
    return (u + 0x7FFFu + ((u >> 16) & 1u)) >> 16;  // RNE
}
DEV unsigned short f2bf(float f) { return (unsigned short)f2bf_bits(f); }

DEV f32x8 zero8() { f32x8 z = {0.f,0.f,0.f,0.f,0.f,0.f,0.f,0.f}; return z; }

DEV f32x8 wmma_bf16(const FragU& a, const FragU& b, f32x8 c) {
    // v_wmma_f32_16x16x32_bf16  D = A(16x32) * B(32x16) + C
    return __builtin_amdgcn_wmma_f32_16x16x32_bf16(
        false, a.v, false, b.v, (short)0, c, false, false);
}

// A-fragment u32 column (within a 32-wide K tile = 16 u32) for vgpr j, half hi.
// ISA 7.12.2: vgpr j<4 holds K = 2j + 8*hi (pairs); j>=4 holds K = 16 + 2(j-4) + 8*hi.
DEV int afrag_col(int j, int hi) { return j + 4*hi + ((j >= 4) ? 4 : 0); }

// Build A fragment from a row-major [16][rowStrideU32] bf16-pair buffer.
DEV FragU load_afrag(const uint32_t* buf, int rowStrideU32, int ktOffU32) {
    const int lane = threadIdx.x & 31;
    const int m = lane & 15, hi = lane >> 4;
    FragU f;
#pragma unroll
    for (int j = 0; j < 8; ++j)
        f.u[j] = buf[m * rowStrideU32 + ktOffU32 + afrag_col(j, hi)];
    return f;
}

// Pre-packed B fragment: [frag][lane][8] u32, contiguous 32B per lane.
DEV FragU load_bfrag(const uint32_t* w, int frag) {
    const int lane = threadIdx.x & 31;
    const uint32_t* p = w + (size_t)frag * 256 + lane * 8;
    FragU f;
#pragma unroll
    for (int j = 0; j < 8; ++j) f.u[j] = p[j];
    return f;
}

DEV float gelu_exact(float v) { return 0.5f * v * (1.f + erff(v * 0.70710678118654752f)); }

// ---------------- packed-weight offsets (u32 units), each frag = 256 u32 ----
#define OFF_F1   0u      //   3x64 : 4 frags
#define OFF_F2   1024u   //  64x3  : 2
#define OFF_PS   1536u   //   3x16 : 1
#define OFF_FP   1792u
#define OFF_Q    2048u
#define OFF_K    2304u
#define OFF_V    2560u
#define OFF_O    2816u
#define OFF_PA   3072u
#define OFF_G1A  3328u   //   3x64 : 4
#define OFF_G1B  4352u   //  64x64 : 8
#define OFF_G2A  6400u   //  64x64 : 8
#define OFF_G2B  8448u   //  64x3  : 2
#define OFF_PG   8960u
#define OFF_PF   9216u   // 128x16 : 4
#define OFF_PB   10240u  //  16x3  : 1
#define OFF_FU   10496u  // 256x256: 128 frags (kt*16+nt)

struct Params {
    const float *x;
    const float *f1_b, *f2_b, *ps_b, *fp_b, *q_b, *k_b, *v_b, *o_b, *pa_b;
    const float *ci_w, *ci_b;
    const float *g1a_b, *g1b_b, *g2a_b, *g2b_b, *pg_b, *pf_b, *pb_b;
    const uint32_t* wpk;   // packed bf16 B-fragments
    const float*    bn;    // [0:256) fu_scale [256:512) fu_shift [512:640) bn2_scale [640:768) bn2_shift
    float*          acc16; // [M][16]  x_self + x_attn + x_neigh
    float*          spec;  // [B][256][N] interleaved re/im spectrum (in-place conv)
    unsigned short* zbf;   // [M][128] bf16 FFCM activation (row-major view)
    float*          out;   // [M][3]
};

struct RawW {
    const float *f1_w,*f2_w,*ps_w,*fp_w,*q_w,*k_w,*v_w,*o_w,*pa_w;
    const float *fu_w,*fu_g,*fu_be,*fu_m,*fu_v,*b2_g,*b2_be,*b2_m,*b2_v;
    const float *g1a_w,*g1b_w,*g2a_w,*g2b_w,*pg_w,*pf_w,*pb_w;
    uint32_t* wpk; float* bn;
};

// ---------------------------------------------------------------------------
// K0: pack fp32 weights into exact CDNA5 WMMA bf16 B-fragment layout.
// B 32x16 layout: lane (hi*16+n) holds column n; vgpr j holds K = kt*32+hi*8+2j+(j>=4?8:0) (+1).
// ---------------------------------------------------------------------------
__device__ void pack_matrix(const float* W, int K, int N, int ldw, bool transposed,
                            uint32_t* dst) {
    const int ktiles = (K + 31) >> 5, ntiles = (N + 15) >> 4;
    const int total = ktiles * ntiles * 32;
    for (int t = threadIdx.x; t < total; t += blockDim.x) {
        const int lane = t & 31, f = t >> 5;
        const int kt = f / ntiles, nt = f - kt * ntiles;
        const int n = nt * 16 + (lane & 15);
        const int hi = lane >> 4;
        uint32_t* o = dst + (size_t)f * 256 + lane * 8;
#pragma unroll
        for (int j = 0; j < 8; ++j) {
            const int k = kt * 32 + hi * 8 + 2 * j + ((j >= 4) ? 8 : 0);
            float lo = 0.f, hv = 0.f;
            if (n < N) {
                if (k     < K) lo = transposed ? W[(size_t)n*ldw + k]     : W[(size_t)k*ldw + n];
                if (k + 1 < K) hv = transposed ? W[(size_t)n*ldw + k + 1] : W[(size_t)(k+1)*ldw + n];
            }
            o[j] = f2bf_bits(lo) | (f2bf_bits(hv) << 16);
        }
    }
}

__global__ __launch_bounds__(256) void k_pack(RawW R) {
    pack_matrix(R.f1_w,   3,  64,  64, false, R.wpk + OFF_F1);
    pack_matrix(R.f2_w,  64,   3,   3, false, R.wpk + OFF_F2);
    pack_matrix(R.ps_w,   3,  16,  16, false, R.wpk + OFF_PS);
    pack_matrix(R.fp_w,   3,  16,  16, false, R.wpk + OFF_FP);
    pack_matrix(R.q_w,   16,  16,  16, false, R.wpk + OFF_Q);
    pack_matrix(R.k_w,   16,  16,  16, false, R.wpk + OFF_K);
    pack_matrix(R.v_w,   16,  16,  16, false, R.wpk + OFF_V);
    pack_matrix(R.o_w,   16,  16,  16, false, R.wpk + OFF_O);
    pack_matrix(R.pa_w,  16,  16,  16, false, R.wpk + OFF_PA);
    pack_matrix(R.g1a_w,  3,  64,  64, false, R.wpk + OFF_G1A);
    pack_matrix(R.g1b_w, 64,  64,  64, false, R.wpk + OFF_G1B);
    pack_matrix(R.g2a_w, 64,  64,  64, false, R.wpk + OFF_G2A);
    pack_matrix(R.g2b_w, 64,   3,   3, false, R.wpk + OFF_G2B);
    pack_matrix(R.pg_w,   3,  16,  16, false, R.wpk + OFF_PG);
    pack_matrix(R.pf_w, 128,  16,  16, false, R.wpk + OFF_PF);
    pack_matrix(R.pb_w,  16,   3,   3, false, R.wpk + OFF_PB);
    pack_matrix(R.fu_w, 256, 256, 256, true,  R.wpk + OFF_FU);  // fu_w is [out][in]
    const int t = threadIdx.x;
    if (t < 256) { float sc = R.fu_g[t] * rsqrtf(R.fu_v[t] + 1e-5f);
                   R.bn[t] = sc; R.bn[256 + t] = R.fu_be[t] - R.fu_m[t] * sc; }
    if (t < 128) { float sc = R.b2_g[t] * rsqrtf(R.b2_v[t] + 1e-5f);
                   R.bn[512 + t] = sc; R.bn[640 + t] = R.b2_be[t] - R.b2_m[t] * sc; }
}

// ---------------------------------------------------------------------------
// K1: fused self + attention + GraphSAGE branches. One wave per 16-node tile.
// ---------------------------------------------------------------------------
__global__ __launch_bounds__(256) void k_branches(Params P) {
    __shared__ uint32_t s_act[8][2][16 * 32];  // per wave: two [16][64]bf16 buffers
    __shared__ float    s_qkv[8][3 * 256];

    const int wave = threadIdx.x >> 5, lane = threadIdx.x & 31;
    const int tile = blockIdx.x * 8 + wave;
    const int R0 = tile * 16;
    const int n = lane & 15, hi = lane >> 4;

    uint32_t* A32 = s_act[wave][0];  unsigned short* A16 = (unsigned short*)A32;
    uint32_t* B32 = s_act[wave][1];  unsigned short* B16 = (unsigned short*)B32;
    float* qf = s_qkv[wave]; float* kf = qf + 256; float* vf = kf + 256;

    auto store_bf = [&](unsigned short* dst, int colOff, f32x8 d, bool relu, bool pad) {
#pragma unroll
        for (int i = 0; i < 8; ++i) {
            const int mm = i + 8 * hi;
            float v = d[i]; if (relu) v = fmaxf(v, 0.f);
            dst[mm * 64 + colOff + n] = f2bf(v);
            if (pad) dst[mm * 64 + colOff + 16 + n] = 0;
        }
    };
    auto store_f32 = [&](float* dst, f32x8 d) {
#pragma unroll
        for (int i = 0; i < 8; ++i) dst[(i + 8 * hi) * 16 + n] = d[i];
    };

    // load x tile [16][3] -> A, K padded to 32
    if (lane < 16) {
        const float* xr = P.x + (size_t)(R0 + lane) * 3;
        A16[lane * 64 + 0] = f2bf(xr[0]);
        A16[lane * 64 + 1] = f2bf(xr[1]);
        A16[lane * 64 + 2] = f2bf(xr[2]);
#pragma unroll
        for (int c = 3; c < 16; ++c) A16[lane * 64 + c] = 0;
    } else {
        const int r = lane - 16;
#pragma unroll
        for (int c = 16; c < 32; ++c) A16[r * 64 + c] = 0;
    }
    const FragU ax = load_afrag(A32, 32, 0);  // x fragment (kept in regs)

    // ---- self branch: relu(x@F1+b) @ F2 + b  ->  @ PS + b
#pragma unroll
    for (int nt = 0; nt < 4; ++nt) {
        f32x8 d = wmma_bf16(ax, load_bfrag(P.wpk + OFF_F1, nt), zero8());
        const float bv = P.f1_b[nt * 16 + n];
#pragma unroll
        for (int i = 0; i < 8; ++i) d[i] += bv;
        store_bf(B16, nt * 16, d, true, false);
    }
    {
        FragU a0 = load_afrag(B32, 32, 0), a1 = load_afrag(B32, 32, 16);
        f32x8 d = wmma_bf16(a0, load_bfrag(P.wpk + OFF_F2, 0), zero8());
        d = wmma_bf16(a1, load_bfrag(P.wpk + OFF_F2, 1), d);
        const float bv = (n < 3) ? P.f2_b[n] : 0.f;
#pragma unroll
        for (int i = 0; i < 8; ++i) d[i] += bv;
        store_bf(A16, 0, d, false, true);
    }
    f32x8 xs = wmma_bf16(load_afrag(A32, 32, 0), load_bfrag(P.wpk + OFF_PS, 0), zero8());
    { const float bv = P.ps_b[n];
#pragma unroll
      for (int i = 0; i < 8; ++i) xs[i] += bv; }

    // ---- attention branch
    {
        f32x8 d = wmma_bf16(ax, load_bfrag(P.wpk + OFF_FP, 0), zero8());
        const float bv = P.fp_b[n];
#pragma unroll
        for (int i = 0; i < 8; ++i) d[i] += bv;
        store_bf(B16, 0, d, false, true);
    }
    const FragU fx = load_afrag(B32, 32, 0);
    { f32x8 d = wmma_bf16(fx, load_bfrag(P.wpk + OFF_Q, 0), zero8());
      const float bv = P.q_b[n];
#pragma unroll
      for (int i = 0; i < 8; ++i) d[i] += bv;  store_f32(qf, d); }
    { f32x8 d = wmma_bf16(fx, load_bfrag(P.wpk + OFF_K, 0), zero8());
      const float bv = P.k_b[n];
#pragma unroll
      for (int i = 0; i < 8; ++i) d[i] += bv;  store_f32(kf, d); }
    { f32x8 d = wmma_bf16(fx, load_bfrag(P.wpk + OFF_V, 0), zero8());
      const float bv = P.v_b[n];
#pragma unroll
      for (int i = 0; i < 8; ++i) d[i] += bv;  store_f32(vf, d); }

    // per-node 4x4 softmax attention (heads mixed within node, per reference)
#pragma unroll
    for (int rr = 0; rr < 2; ++rr) {
        const int r = lane + rr * 32;            // 64 (node,head) rows / 32 lanes
        const int node = r >> 2, h = r & 3;
        const float* qp = qf + node * 16 + h * 4;
        float q0 = qp[0], q1 = qp[1], q2 = qp[2], q3 = qp[3];
        float s[4], mx = -1e30f;
#pragma unroll
        for (int g = 0; g < 4; ++g) {
            const float* kp = kf + node * 16 + g * 4;
            s[g] = 0.5f * (q0 * kp[0] + q1 * kp[1] + q2 * kp[2] + q3 * kp[3]);
            mx = fmaxf(mx, s[g]);
        }
        float e[4], sum = 0.f;
#pragma unroll
        for (int g = 0; g < 4; ++g) { e[g] = expf(s[g] - mx); sum += e[g]; }
        const float inv = 1.f / sum;
#pragma unroll
        for (int d2 = 0; d2 < 4; ++d2) {
            float o = 0.f;
#pragma unroll
            for (int g = 0; g < 4; ++g) o += e[g] * vf[node * 16 + g * 4 + d2];
            A16[node * 64 + h * 4 + d2] = f2bf(o * inv);
        }
    }
#pragma unroll
    for (int i = 0; i < 8; ++i) A16[(i + 8 * hi) * 64 + 16 + n] = 0;  // zero K pad

    {
        f32x8 d = wmma_bf16(load_afrag(A32, 32, 0), load_bfrag(P.wpk + OFF_O, 0), zero8());
        const float bv = P.o_b[n];
#pragma unroll
        for (int i = 0; i < 8; ++i) d[i] += bv;
        store_bf(B16, 0, d, false, true);
    }
    f32x8 xa = wmma_bf16(load_afrag(B32, 32, 0), load_bfrag(P.wpk + OFF_PA, 0), zero8());
    { const float bv = P.pa_b[n];
#pragma unroll
      for (int i = 0; i < 8; ++i) xa[i] += bv; }

    // ---- GraphSAGE branch (adj=None => MLP): relu(x@G1A+b)@G1B+b ; relu(.@G2A+b)@G2B+b ; @PG+b
#pragma unroll
    for (int nt = 0; nt < 4; ++nt) {
        f32x8 d = wmma_bf16(ax, load_bfrag(P.wpk + OFF_G1A, nt), zero8());
        const float bv = P.g1a_b[nt * 16 + n];
#pragma unroll
        for (int i = 0; i < 8; ++i) d[i] += bv;
        store_bf(B16, nt * 16, d, true, false);
    }
    {
        FragU a0 = load_afrag(B32, 32, 0), a1 = load_afrag(B32, 32, 16);
#pragma unroll
        for (int nt = 0; nt < 4; ++nt) {
            f32x8 d = wmma_bf16(a0, load_bfrag(P.wpk + OFF_G1B, nt), zero8());
            d = wmma_bf16(a1, load_bfrag(P.wpk + OFF_G1B, 4 + nt), d);
            const float bv = P.g1b_b[nt * 16 + n];
#pragma unroll
            for (int i = 0; i < 8; ++i) d[i] += bv;
            store_bf(A16, nt * 16, d, false, false);
        }
    }
    {
        FragU a0 = load_afrag(A32, 32, 0), a1 = load_afrag(A32, 32, 16);
#pragma unroll
        for (int nt = 0; nt < 4; ++nt) {
            f32x8 d = wmma_bf16(a0, load_bfrag(P.wpk + OFF_G2A, nt), zero8());
            d = wmma_bf16(a1, load_bfrag(P.wpk + OFF_G2A, 4 + nt), d);
            const float bv = P.g2a_b[nt * 16 + n];
#pragma unroll
            for (int i = 0; i < 8; ++i) d[i] += bv;
            store_bf(B16, nt * 16, d, true, false);
        }
    }
    {
        FragU a0 = load_afrag(B32, 32, 0), a1 = load_afrag(B32, 32, 16);
        f32x8 d = wmma_bf16(a0, load_bfrag(P.wpk + OFF_G2B, 0), zero8());
        d = wmma_bf16(a1, load_bfrag(P.wpk + OFF_G2B, 1), d);
        const float bv = (n < 3) ? P.g2b_b[n] : 0.f;
#pragma unroll
        for (int i = 0; i < 8; ++i) d[i] += bv;
        store_bf(A16, 0, d, false, true);
    }
    f32x8 xg = wmma_bf16(load_afrag(A32, 32, 0), load_bfrag(P.wpk + OFF_PG, 0), zero8());
    { const float bv = P.pg_b[n];
#pragma unroll
      for (int i = 0; i < 8; ++i) xg[i] += bv; }

#pragma unroll
    for (int i = 0; i < 8; ++i)
        P.acc16[(size_t)(R0 + i + 8 * hi) * 16 + n] = xs[i] + xa[i] + xg[i];
}

// ---------------------------------------------------------------------------
// K2/K4: length-8192 Stockham FFT in LDS (ping-pong, 128KB dynamic LDS).
// Forward: fuses xi = gelu(conv_init(x)) generation; writes ortho spectrum.
// Inverse: fuses residual + BN2 + ReLU, writes bf16 rows for final GEMM.
// One workgroup (256 thr) per (batch, channel) pair: grid = 16*128 = 2048.
// ---------------------------------------------------------------------------
__global__ __launch_bounds__(256) void k_fft(Params P, int inverse) {
    extern __shared__ float smem[];
    float* rA = smem;          float* iA = smem + 8192;
    float* rB = smem + 16384;  float* iB = smem + 24576;

    const int bb = blockIdx.x >> 7;
    const int c  = blockIdx.x & 127;
    const float w0 = P.ci_w[c * 64 + 0], w1 = P.ci_w[c * 64 + 1],
                w2 = P.ci_w[c * 64 + 2], cb = P.ci_b[c];

    if (!inverse) {
        for (int t = threadIdx.x; t < 8192; t += 256) {
            const float* xr = P.x + ((size_t)bb * 8192 + t) * 3;
            float v = fmaf(w2, xr[2], fmaf(w1, xr[1], fmaf(w0, xr[0], cb)));
            rA[t] = gelu_exact(v); iA[t] = 0.f;
        }
    } else {
        const float* sR = P.spec + ((size_t)bb * 256 + 2 * c) * 8192;
        const float* sI = sR + 8192;
        for (int t = threadIdx.x; t < 8192; t += 256) { rA[t] = sR[t]; iA[t] = sI[t]; }
    }
    __syncthreads();

    float *sr = rA, *si = iA, *dr = rB, *di = iB;
    const float sgn = inverse ? 3.14159265358979323846f : -3.14159265358979323846f;
    for (int s = 0; s < 13; ++s) {                  // radix-2 Stockham, self-sorting
        const int m = 1 << s;
        const float invl = 1.f / (float)(4096 >> s);
        for (int it = 0; it < 16; ++it) {
            const int idx = threadIdx.x + (it << 8);
            const int k = idx & (m - 1);
            const int j = idx >> s;
            float sw, cw; __sincosf(sgn * (float)j * invl, &sw, &cw);
            const int ia = k + j * m, ib = ia + 4096;   // l*m == 4096 every stage
            const float ar = sr[ia], ai = si[ia], br = sr[ib], bi = si[ib];
            const float qr = ar - br, qi = ai - bi;
            const int o0 = k + 2 * j * m;
            dr[o0] = ar + br;            di[o0] = ai + bi;
            dr[o0 + m] = qr * cw - qi * sw;  di[o0 + m] = qr * sw + qi * cw;
        }
        __syncthreads();
        float* t = sr; sr = dr; dr = t;  t = si; si = di; di = t;
    }

    const float scale = 0.011048543456039805f;  // 1/sqrt(8192)  (ortho)
    if (!inverse) {
        float* oR = P.spec + ((size_t)bb * 256 + 2 * c) * 8192;
        float* oI = oR + 8192;
        for (int t = threadIdx.x; t < 8192; t += 256) { oR[t] = sr[t] * scale; oI[t] = si[t] * scale; }
    } else {
        const float sc2 = P.bn[512 + c], sh2 = P.bn[640 + c];
        unsigned short* z = P.zbf + (size_t)bb * 1048576 + (size_t)c * 8192;
        for (int t = threadIdx.x; t < 8192; t += 256) {
            const float* xr = P.x + ((size_t)bb * 8192 + t) * 3;
            float v = fmaf(w2, xr[2], fmaf(w1, xr[1], fmaf(w0, xr[0], cb)));
            const float xi = gelu_exact(v);                        // recomputed residual
            const float sum = sr[t] * scale + xi;                  // real(ifft) + xi
            z[t] = f2bf(fmaxf(fmaf(sum, sc2, sh2), 0.f));          // BN2 + ReLU -> bf16
        }
    }
}

// ---------------------------------------------------------------------------
// K3: FourierUnit per-bin conv 256->256 + BN + ReLU, in-place on spectrum.
// One wave per (batch, 16-bin tile): A[16 bins][256 ch] in LDS, 8x16 WMMAs.
// ---------------------------------------------------------------------------
__global__ __launch_bounds__(256) void k_fuconv(Params P) {
    __shared__ uint32_t sA[8][16 * 128];   // per wave [16][256] bf16
    const int wave = threadIdx.x >> 5, lane = threadIdx.x & 31;
    const int tile = blockIdx.x * 8 + wave;
    const int bb = tile >> 9;
    const int n0 = (tile & 511) << 4;
    uint32_t* A32 = sA[wave];
    unsigned short* A16 = (unsigned short*)A32;
    const int n = lane & 15, hi = lane >> 4;

    __builtin_prefetch((const void*)(P.wpk + OFF_FU), 0, 3);  // global_prefetch_b8

    // gather A tile: lane = channel within 32-group, 16 consecutive bins each
#pragma unroll
    for (int cc = 0; cc < 8; ++cc) {
        const int c = cc * 32 + lane;
        const float4* s4 = (const float4*)(P.spec + ((size_t)bb * 256 + c) * 8192 + n0);
#pragma unroll
        for (int t4 = 0; t4 < 4; ++t4) {
            const float4 v = s4[t4];
            A16[(t4 * 4 + 0) * 256 + c] = f2bf(v.x);
            A16[(t4 * 4 + 1) * 256 + c] = f2bf(v.y);
            A16[(t4 * 4 + 2) * 256 + c] = f2bf(v.z);
            A16[(t4 * 4 + 3) * 256 + c] = f2bf(v.w);
        }
    }

#pragma unroll 1
    for (int nt = 0; nt < 16; ++nt) {
        f32x8 acc = zero8();
#pragma unroll
        for (int kt = 0; kt < 8; ++kt)
            acc = wmma_bf16(load_afrag(A32, 128, kt * 16),
                            load_bfrag(P.wpk + OFF_FU, kt * 16 + nt), acc);
        const int o = nt * 16 + n;
        const float sc = P.bn[o], sh = P.bn[256 + o];
        float* dst = P.spec + ((size_t)bb * 256 + o) * 8192 + n0 + hi * 8;
#pragma unroll
        for (int i = 0; i < 8; ++i) dst[i] = fmaxf(fmaf(acc[i], sc, sh), 0.f);
    }
}

// ---------------------------------------------------------------------------
// K5: out = ((acc16 + zrow@PF + pf_b) @ PB) + pb_b   (16 rows per wave)
// ---------------------------------------------------------------------------
__global__ __launch_bounds__(256) void k_final(Params P) {
    __shared__ uint32_t sT[8][16 * 16];
    const int wave = threadIdx.x >> 5, lane = threadIdx.x & 31;
    const int tile = blockIdx.x * 8 + wave;
    const int R0 = tile * 16;
    const int n = lane & 15, hi = lane >> 4;
    uint32_t* T32 = sT[wave];
    unsigned short* T16 = (unsigned short*)T32;
    const uint32_t* z32 = (const uint32_t*)P.zbf;   // row-major [M][64] u32 (bf16 pairs)

    f32x8 acc = zero8();
#pragma unroll
    for (int kt = 0; kt < 4; ++kt) {
        FragU a;
#pragma unroll
        for (int j = 0; j < 8; ++j)
            a.u[j] = z32[(size_t)(R0 + n) * 64 + kt * 16 + afrag_col(j, hi)];
        acc = wmma_bf16(a, load_bfrag(P.wpk + OFF_PF, kt), acc);
    }
    const float bv = P.pf_b[n];
#pragma unroll
    for (int i = 0; i < 8; ++i) {
        acc[i] += bv + P.acc16[(size_t)(R0 + i + 8 * hi) * 16 + n];
        T16[(i + 8 * hi) * 32 + n] = f2bf(acc[i]);
        T16[(i + 8 * hi) * 32 + 16 + n] = 0;
    }
    f32x8 o = wmma_bf16(load_afrag(T32, 16, 0), load_bfrag(P.wpk + OFF_PB, 0), zero8());
    if (n < 3) {
        const float ob = P.pb_b[n];
#pragma unroll
        for (int i = 0; i < 8; ++i)
            P.out[(size_t)(R0 + i + 8 * hi) * 3 + n] = o[i] + ob;
    }
}

// ---------------------------------------------------------------------------
extern "C" void kernel_launch(void* const* d_in, const int* in_sizes, int n_in,
                              void* d_out, int out_size, void* d_ws, size_t ws_size,
                              hipStream_t stream) {
    (void)in_sizes; (void)n_in; (void)out_size; (void)ws_size;
    char* ws = (char*)d_ws;
    uint32_t*       wpk   = (uint32_t*)(ws);                      // ~170KB packed weights
    float*          bn    = (float*)(ws + (1ull << 20));          // BN affines
    float*          acc16 = (float*)(ws + (2ull << 20));          // 8MB
    float*          spec  = (float*)(ws + (16ull << 20));         // 128MB spectrum
    unsigned short* zbf   = (unsigned short*)(ws + (144ull << 20)); // 32MB

    const float* in[44];
    for (int i = 0; i < 44; ++i) in[i] = (const float*)d_in[i];

    RawW R;
    R.f1_w = in[1];  R.f2_w = in[3];  R.ps_w = in[5];  R.fp_w = in[7];
    R.q_w  = in[9];  R.k_w  = in[11]; R.v_w  = in[13]; R.o_w  = in[15]; R.pa_w = in[17];
    R.fu_w = in[21];
    R.fu_g = in[22]; R.fu_be = in[23]; R.fu_m = in[24]; R.fu_v = in[25];
    R.b2_g = in[26]; R.b2_be = in[27]; R.b2_m = in[28]; R.b2_v = in[29];
    R.g1a_w = in[32]; R.g1b_w = in[34]; R.g2a_w = in[36]; R.g2b_w = in[38];
    R.pg_w = in[40]; R.pf_w = in[30]; R.pb_w = in[42];
    R.wpk = wpk; R.bn = bn;

    Params P;
    P.x = in[0];
    P.f1_b = in[2]; P.f2_b = in[4]; P.ps_b = in[6]; P.fp_b = in[8];
    P.q_b = in[10]; P.k_b = in[12]; P.v_b = in[14]; P.o_b = in[16]; P.pa_b = in[18];
    P.ci_w = in[19]; P.ci_b = in[20];
    P.g1a_b = in[33]; P.g1b_b = in[35]; P.g2a_b = in[37]; P.g2b_b = in[39];
    P.pg_b = in[41]; P.pf_b = in[31]; P.pb_b = in[43];
    P.wpk = wpk; P.bn = bn; P.acc16 = acc16; P.spec = spec; P.zbf = zbf;
    P.out = (float*)d_out;

    k_pack    <<<1,    256, 0,      stream>>>(R);
    k_branches<<<1024, 256, 0,      stream>>>(P);      // 8192 tiles / 8 waves
    k_fft     <<<2048, 256, 131072, stream>>>(P, 0);   // forward FFT per (b,c)
    k_fuconv  <<<1024, 256, 0,      stream>>>(P);      // per-bin 256x256 WMMA conv
    k_fft     <<<2048, 256, 131072, stream>>>(P, 1);   // inverse FFT + residual + BN2
    k_final   <<<1024, 256, 0,      stream>>>(P);      // 128->16, sum, 16->3
}
